// PointNetPlusPlus_63093069578691
// MI455X (gfx1250) — compile-verified, matrix operand order checked by reading, not run on hardware
//
#include <hip/hip_runtime.h>
#include <hip/hip_bf16.h>

// PointNet++ segmentation forward for MI455X (gfx1250, wave32).
// GEMMs: f16 WMMA (v_wmma_f32_16x16x32_f16), one wave = one 16x16 tile.
// A/B tiles staged to LDS with global_load_async_to_lds_b128 (double-buffered,
// pipelined via s_wait_asynccnt). Weights/activations are pre-converted to
// zero-padded f16 (K -> x32, O -> x16) so the WMMA loop is branch-free and
// EXEC stays all-ones (ISA requirement for WMMA).

typedef __attribute__((ext_vector_type(16))) _Float16 v16h;
typedef __attribute__((ext_vector_type(8)))  _Float16 v8h;
typedef __attribute__((ext_vector_type(8)))  float    v8f;

#define EPS_BN 1e-5f

// ---------------------------------------------------------------------------
// Input split/transpose: xyz [B,6,N] -> xyz_t [B,N,3], pts_t [B,N,6]
// ---------------------------------------------------------------------------
__global__ __launch_bounds__(256) void split_input_kernel(
    const float* __restrict__ xyz, float* __restrict__ xyz_t,
    float* __restrict__ pts_t, int B, int N) {
  int i = blockIdx.x * blockDim.x + threadIdx.x;
  if (i >= B * N) return;
  int b = i / N, n = i % N;
  #pragma unroll
  for (int c = 0; c < 3; ++c)
    xyz_t[(size_t)i * 3 + c] = xyz[((size_t)b * 6 + c) * N + n];
  #pragma unroll
  for (int c = 0; c < 6; ++c)
    pts_t[(size_t)i * 6 + c] = xyz[((size_t)b * 6 + c) * N + n];
}

// ---------------------------------------------------------------------------
// Farthest point sampling: one block per batch, distance array in LDS.
// ---------------------------------------------------------------------------
__global__ __launch_bounds__(256) void fps_kernel(
    const float* __restrict__ xyz_t, int* __restrict__ fps_idx,
    int N, int npoint) {
  __shared__ float dist[4096];
  __shared__ float rv[256];
  __shared__ int   ri[256];
  __shared__ int   far_s;
  const int b = blockIdx.x;
  const int tid = threadIdx.x;
  const float* P = xyz_t + (size_t)b * N * 3;

  for (int j = tid; j < N; j += 256) dist[j] = 1e10f;
  if (tid == 0) far_s = 0;
  __syncthreads();

  for (int it = 0; it < npoint; ++it) {
    const int far = far_s;
    if (tid == 0) fps_idx[b * npoint + it] = far;
    const float cx = P[far * 3 + 0];
    const float cy = P[far * 3 + 1];
    const float cz = P[far * 3 + 2];
    float best = -1.0f; int besti = 0;
    for (int j = tid; j < N; j += 256) {
      float dx = P[j * 3 + 0] - cx;
      float dy = P[j * 3 + 1] - cy;
      float dz = P[j * 3 + 2] - cz;
      float d  = dx * dx + dy * dy + dz * dz;
      float nd = fminf(dist[j], d);
      dist[j] = nd;
      if (nd > best) { best = nd; besti = j; }
    }
    rv[tid] = best; ri[tid] = besti;
    __syncthreads();
    for (int s = 128; s > 0; s >>= 1) {
      if (tid < s && rv[tid + s] > rv[tid]) { rv[tid] = rv[tid + s]; ri[tid] = ri[tid + s]; }
      __syncthreads();
    }
    if (tid == 0) far_s = ri[0];
    __syncthreads();
  }
}

__global__ __launch_bounds__(256) void gather_new_xyz_kernel(
    const float* __restrict__ xyz_t, const int* __restrict__ fps_idx,
    float* __restrict__ new_xyz, int B, int N, int S) {
  int i = blockIdx.x * blockDim.x + threadIdx.x;
  if (i >= B * S) return;
  int b = i / S;
  int id = fps_idx[i];
  #pragma unroll
  for (int c = 0; c < 3; ++c)
    new_xyz[(size_t)i * 3 + c] = xyz_t[((size_t)b * N + id) * 3 + c];
}

__global__ __launch_bounds__(256) void ball_query_kernel(
    const float* __restrict__ xyz_t, const float* __restrict__ new_xyz,
    int* __restrict__ idx, int B, int N, int S, float r2, int ns) {
  int i = blockIdx.x * blockDim.x + threadIdx.x;
  if (i >= B * S) return;
  int b = i / S;
  const float* P = xyz_t + (size_t)b * N * 3;
  const float cx = new_xyz[(size_t)i * 3 + 0];
  const float cy = new_xyz[(size_t)i * 3 + 1];
  const float cz = new_xyz[(size_t)i * 3 + 2];
  int* out = idx + (size_t)i * ns;
  int cnt = 0, first = 0;
  bool have_first = false;
  for (int j = 0; j < N && cnt < ns; ++j) {
    float dx = P[j * 3 + 0] - cx;
    float dy = P[j * 3 + 1] - cy;
    float dz = P[j * 3 + 2] - cz;
    float d2 = dx * dx + dy * dy + dz * dz;
    if (d2 <= r2) {
      if (!have_first) { first = j; have_first = true; }
      out[cnt++] = j;
    }
  }
  for (; cnt < ns; ++cnt) out[cnt] = first;
}

__global__ __launch_bounds__(256) void group_features_kernel(
    const float* __restrict__ xyz_t, const float* __restrict__ pts_t,
    const float* __restrict__ new_xyz, const int* __restrict__ idx,
    float* __restrict__ X, int B, int N, int S, int ns, int Cp) {
  int row = blockIdx.x * blockDim.x + threadIdx.x;
  int M = B * S * ns;
  if (row >= M) return;
  int bs = row / ns;
  int b  = bs / S;
  int id = idx[row];
  const float* p = xyz_t + ((size_t)b * N + id) * 3;
  const float* c = new_xyz + (size_t)bs * 3;
  float* xr = X + (size_t)row * (3 + Cp);
  xr[0] = p[0] - c[0];
  xr[1] = p[1] - c[1];
  xr[2] = p[2] - c[2];
  const float* f = pts_t + ((size_t)b * N + id) * Cp;
  for (int cc = 0; cc < Cp; ++cc) xr[3 + cc] = f[cc];
}

// ---------------------------------------------------------------------------
// f32 -> padded f16 conversions (zero-pad K to Kpad; weights also pad O).
// ---------------------------------------------------------------------------
__global__ __launch_bounds__(256) void cvt_x_kernel(
    const float* __restrict__ X, _Float16* __restrict__ Xh,
    int M, int K, int Kpad) {
  int i = blockIdx.x * blockDim.x + threadIdx.x;
  if (i >= M * Kpad) return;
  int m = i / Kpad, k = i % Kpad;
  Xh[i] = (k < K) ? (_Float16)X[(size_t)m * K + k] : (_Float16)0.0f;
}

__global__ __launch_bounds__(256) void cvt_w_kernel(
    const float* __restrict__ W, const float* __restrict__ b,
    _Float16* __restrict__ Wh, float* __restrict__ bp,
    int O, int K, int Opad, int Kpad) {
  int i = blockIdx.x * blockDim.x + threadIdx.x;
  if (i >= Opad * Kpad) return;
  int o = i / Kpad, k = i % Kpad;
  Wh[i] = (o < O && k < K) ? (_Float16)W[(size_t)o * K + k] : (_Float16)0.0f;
  if (k == 0) bp[o] = (o < O) ? b[o] : 0.0f;
}

// ---------------------------------------------------------------------------
// WMMA GEMM: Y[m, o] = sum_k Xh[m,k] * Wh[o,k] + bp[o]
//   Xh: [M, Kpad] f16 (M % 16 == 0, Kpad % 32 == 0, zero-padded)
//   Wh: [Opad, Kpad] f16, bp: [Opad] f32, Y: [M, Opad] f32
// One wave per 16x16 tile. Per K-step each wave stages its 16x32 A and B
// tiles (1 KB each) into per-wave LDS buffers with
// global_load_async_to_lds_b128, double-buffered; fragments are then two
// aligned ds_load_b128 each, matching the ISA f16 A/B VGPR layout:
//   VGPR0..3 halves -> K = 8*khalf + 0..7, VGPR4..7 -> K = 16 + 8*khalf + 0..7.
// All control flow is wave-uniform -> EXEC all-ones at every WMMA.
// ---------------------------------------------------------------------------
__global__ __launch_bounds__(128) void gemm_wmma_kernel(
    const _Float16* __restrict__ Xh, const _Float16* __restrict__ Wh,
    const float* __restrict__ bp, float* __restrict__ Y,
    int M, int Kpad, int Opad) {
  // [wave][buf][A|B][512 halves]  = 16 KB
  __shared__ __align__(16) _Float16 stage[4][2][2][512];

  const int lane  = threadIdx.x & 31;
  const int wave  = threadIdx.x >> 5;
  const int tilesO = Opad >> 4;
  const int tilesM = M >> 4;
  const int tileId = blockIdx.x * 4 + wave;   // wave-uniform
  if (tileId >= tilesM * tilesO) return;      // wave-uniform exit
  const int tM = tileId / tilesO;
  const int tO = tileId % tilesO;

  const int mlo   = lane & 15;
  const int khalf = lane >> 4;

  const _Float16* arow0 = Xh + (size_t)(tM * 16) * Kpad;
  const _Float16* brow0 = Wh + (size_t)(tO * 16) * Kpad;

  v8f acc;
  const float bv = bp[tO * 16 + mlo];
  #pragma unroll
  for (int r = 0; r < 8; ++r) acc[r] = bv;

  // Stage one K-step (16x32 halves per tile = 64 16B chunks; 2 chunks/lane).
  auto stage_k = [&](int k0, int buf) {
    _Float16* aT = &stage[wave][buf][0][0];
    _Float16* bT = &stage[wave][buf][1][0];
    #pragma unroll
    for (int t = 0; t < 2; ++t) {
      const int c   = lane * 2 + t;      // chunk id 0..63
      const int row = c >> 2;            // tile row (A: m, B: o)
      const int sub = c & 3;             // 16B sub-chunk within the row
      const _Float16* ga = arow0 + (size_t)row * Kpad + k0 + sub * 8;
      const _Float16* gb = brow0 + (size_t)row * Kpad + k0 + sub * 8;
      const unsigned la = (unsigned)(uintptr_t)(aT + c * 8);
      const unsigned lb = (unsigned)(uintptr_t)(bT + c * 8);
      asm volatile("global_load_async_to_lds_b128 %0, %1, off"
                   :: "v"(la), "v"((unsigned long long)(uintptr_t)ga) : "memory");
      asm volatile("global_load_async_to_lds_b128 %0, %1, off"
                   :: "v"(lb), "v"((unsigned long long)(uintptr_t)gb) : "memory");
    }
  };

  stage_k(0, 0);
  for (int k0 = 0; k0 < Kpad; k0 += 32) {
    const int buf = (k0 >> 5) & 1;
    if (k0 + 32 < Kpad) {
      stage_k(k0 + 32, buf ^ 1);
      // async loads retire in order: <=4 outstanding => current buffer done
      asm volatile("s_wait_asynccnt 0x4" ::: "memory");
    } else {
      asm volatile("s_wait_asynccnt 0x0" ::: "memory");
    }
    const _Float16* aT = &stage[wave][buf][0][0];
    const _Float16* bT = &stage[wave][buf][1][0];
    v8h a0 = *(const v8h*)(aT + mlo * 32 + khalf * 8);
    v8h a1 = *(const v8h*)(aT + mlo * 32 + 16 + khalf * 8);
    v8h b0 = *(const v8h*)(bT + mlo * 32 + khalf * 8);
    v8h b1 = *(const v8h*)(bT + mlo * 32 + 16 + khalf * 8);
    v16h a = __builtin_shufflevector(a0, a1, 0,1,2,3,4,5,6,7,8,9,10,11,12,13,14,15);
    v16h b = __builtin_shufflevector(b0, b1, 0,1,2,3,4,5,6,7,8,9,10,11,12,13,14,15);
    acc = __builtin_amdgcn_wmma_f32_16x16x32_f16(
        /*neg_a=*/false, a, /*neg_b=*/false, b,
        /*c_mod=*/(short)0, acc, /*reuse_a=*/false, /*reuse_b=*/false);
  }

  const int n = tO * 16 + mlo;
  #pragma unroll
  for (int r = 0; r < 8; ++r)
    Y[(size_t)(tM * 16 + r + khalf * 8) * Opad + n] = acc[r];
}

// ---------------------------------------------------------------------------
// BatchNorm statistics + apply (ld == O for all BN layers by construction).
// ---------------------------------------------------------------------------
__global__ __launch_bounds__(256) void bn_stats_kernel(
    const float* __restrict__ Y, float* __restrict__ mean,
    float* __restrict__ inv_std, int M, int O) {
  __shared__ float s1[256], s2[256];
  const int o = blockIdx.x;
  const int tid = threadIdx.x;
  float s = 0.0f, ss = 0.0f;
  for (int m = tid; m < M; m += 256) {
    float v = Y[(size_t)m * O + o];
    s += v; ss += v * v;
  }
  s1[tid] = s; s2[tid] = ss;
  __syncthreads();
  for (int st = 128; st > 0; st >>= 1) {
    if (tid < st) { s1[tid] += s1[tid + st]; s2[tid] += s2[tid + st]; }
    __syncthreads();
  }
  if (tid == 0) {
    float mu  = s1[0] / (float)M;
    float var = s2[0] / (float)M - mu * mu;
    mean[o]    = mu;
    inv_std[o] = rsqrtf(var + EPS_BN);
  }
}

__global__ __launch_bounds__(256) void bn_relu_kernel(
    float* __restrict__ Y, const float* __restrict__ mean,
    const float* __restrict__ inv_std, const float* __restrict__ gamma,
    const float* __restrict__ beta, int M, int O) {
  int i = blockIdx.x * blockDim.x + threadIdx.x;
  if (i >= M * O) return;
  int o = i % O;
  float v = (Y[i] - mean[o]) * inv_std[o] * gamma[o] + beta[o];
  Y[i] = fmaxf(v, 0.0f);
}

__global__ __launch_bounds__(256) void maxpool_kernel(
    const float* __restrict__ Y, float* __restrict__ P, int S, int ns, int O) {
  int i = blockIdx.x * blockDim.x + threadIdx.x;
  if (i >= S * O) return;
  int s = i / O, o = i % O;
  float m = -1e30f;
  for (int k = 0; k < ns; ++k)
    m = fmaxf(m, Y[((size_t)s * ns + k) * O + o]);
  P[i] = m;
}

// ---------------------------------------------------------------------------
// 3-NN inverse distance interpolation (feature propagation).
// ---------------------------------------------------------------------------
__global__ __launch_bounds__(256) void fp_interp_kernel(
    const float* __restrict__ xyz1, const float* __restrict__ xyz2,
    const float* __restrict__ p2, float* __restrict__ outI,
    int B, int n1, int n2, int C) {
  int i = blockIdx.x * blockDim.x + threadIdx.x;
  if (i >= B * n1) return;
  int b = i / n1;
  const float* q = xyz1 + (size_t)i * 3;
  const float* P = xyz2 + (size_t)b * n2 * 3;
  float d0 = 1e30f, d1 = 1e30f, d2 = 1e30f;
  int i0 = 0, i1 = 0, i2 = 0;
  for (int j = 0; j < n2; ++j) {
    float dx = P[j * 3 + 0] - q[0];
    float dy = P[j * 3 + 1] - q[1];
    float dz = P[j * 3 + 2] - q[2];
    float d  = dx * dx + dy * dy + dz * dz;
    if (d < d0)      { d2 = d1; i2 = i1; d1 = d0; i1 = i0; d0 = d; i0 = j; }
    else if (d < d1) { d2 = d1; i2 = i1; d1 = d;  i1 = j; }
    else if (d < d2) { d2 = d;  i2 = j; }
  }
  float w0 = 1.0f / (d0 + 1e-8f);
  float w1 = 1.0f / (d1 + 1e-8f);
  float w2 = 1.0f / (d2 + 1e-8f);
  float ws = w0 + w1 + w2;
  w0 /= ws; w1 /= ws; w2 /= ws;
  const float* r0 = p2 + ((size_t)b * n2 + i0) * C;
  const float* r1 = p2 + ((size_t)b * n2 + i1) * C;
  const float* r2 = p2 + ((size_t)b * n2 + i2) * C;
  float* o = outI + (size_t)i * C;
  for (int c = 0; c < C; ++c) o[c] = w0 * r0[c] + w1 * r1[c] + w2 * r2[c];
}

__global__ __launch_bounds__(256) void concat_rows_kernel(
    const float* __restrict__ A, const float* __restrict__ Bm,
    float* __restrict__ X, int M, int Ca, int Cb) {
  int i = blockIdx.x * blockDim.x + threadIdx.x;
  int C = Ca + Cb;
  if (i >= M * C) return;
  int r = i / C, c = i % C;
  X[i] = (c < Ca) ? A[(size_t)r * Ca + c] : Bm[(size_t)r * Cb + (c - Ca)];
}

// ---------------------------------------------------------------------------
// log_softmax over channels (row stride ld); write transposed to [B, O, N]
// ---------------------------------------------------------------------------
__global__ __launch_bounds__(256) void logsoftmax_kernel(
    const float* __restrict__ Y, float* __restrict__ out,
    int B, int N, int O, int ld) {
  int i = blockIdx.x * blockDim.x + threadIdx.x;
  if (i >= B * N) return;
  int b = i / N, n = i % N;
  const float* y = Y + (size_t)i * ld;
  float mx = -1e30f;
  for (int o = 0; o < O; ++o) mx = fmaxf(mx, y[o]);
  float s = 0.0f;
  for (int o = 0; o < O; ++o) s += expf(y[o] - mx);
  float lse = mx + logf(s);
  for (int o = 0; o < O; ++o)
    out[((size_t)b * O + o) * N + n] = y[o] - lse;
}

// ===========================================================================
// Host-side orchestration
// ===========================================================================
namespace {

struct Prepped {
  const _Float16* Wh;
  const float* bp;
  const float* g;
  const float* be;
  int K, O, Kpad, Opad;
};

inline int ceil_div(int a, int b) { return (a + b - 1) / b; }

void run_mlp_layer(const float* X, const Prepped& p, float* Y, _Float16* Xh,
                   float* mean, float* invstd, int M, hipStream_t stream,
                   bool bn_relu = true) {
  cvt_x_kernel<<<ceil_div(M * p.Kpad, 256), 256, 0, stream>>>(X, Xh, M, p.K, p.Kpad);
  const int tiles = (M >> 4) * (p.Opad >> 4);
  gemm_wmma_kernel<<<ceil_div(tiles, 4), 128, 0, stream>>>(
      Xh, p.Wh, p.bp, Y, M, p.Kpad, p.Opad);
  if (bn_relu) {  // Opad == O for every BN layer in this network
    bn_stats_kernel<<<p.O, 256, 0, stream>>>(Y, mean, invstd, M, p.O);
    const int total = M * p.O;
    bn_relu_kernel<<<ceil_div(total, 256), 256, 0, stream>>>(
        Y, mean, invstd, p.g, p.be, M, p.O);
  }
}

void run_sa(const float* in_xyz, const float* in_pts, int Nin, int Cp,
            int S, float radius, const Prepped* mlp, int nl,
            float* out_xyz, float* out_pts,
            int* fpsidx, int* ballidx, float* bufA, float* bufB, _Float16* Xh,
            float* mean, float* invstd, int B, hipStream_t stream) {
  const int ns = 32;
  fps_kernel<<<B, 256, 0, stream>>>(in_xyz, fpsidx, Nin, S);
  const int BS = B * S;
  gather_new_xyz_kernel<<<ceil_div(BS, 256), 256, 0, stream>>>(
      in_xyz, fpsidx, out_xyz, B, Nin, S);
  ball_query_kernel<<<ceil_div(BS, 256), 256, 0, stream>>>(
      in_xyz, out_xyz, ballidx, B, Nin, S, radius * radius, ns);
  const int M = BS * ns;
  group_features_kernel<<<ceil_div(M, 256), 256, 0, stream>>>(
      in_xyz, in_pts, out_xyz, ballidx, bufA, B, Nin, S, ns, Cp);
  const float* X = bufA;
  float* Y = bufB;
  int O = 0;
  for (int l = 0; l < nl; ++l) {
    run_mlp_layer(X, mlp[l], Y, Xh, mean, invstd, M, stream);
    O = mlp[l].O;
    const float* t = X; X = Y; Y = const_cast<float*>(t);
  }
  maxpool_kernel<<<ceil_div(BS * O, 256), 256, 0, stream>>>(X, out_pts, BS, ns, O);
}

void run_fp(const float* xyz1, const float* xyz2,
            const float* pts1, int C1, const float* pts2, int C2,
            int n1, int n2, const Prepped* mlp, int nl,
            float* out_pts, float* bufA, float* bufB, _Float16* Xh,
            float* mean, float* invstd, int B, hipStream_t stream) {
  const int M = B * n1;
  fp_interp_kernel<<<ceil_div(M, 256), 256, 0, stream>>>(
      xyz1, xyz2, pts2, bufA, B, n1, n2, C2);
  const float* X;
  if (C1 > 0) {
    concat_rows_kernel<<<ceil_div(M * (C1 + C2), 256), 256, 0, stream>>>(
        pts1, bufA, bufB, M, C1, C2);
    X = bufB;
  } else {
    X = bufA;
  }
  for (int l = 0; l < nl; ++l) {
    float* dst = (l == nl - 1) ? out_pts : ((X == bufA) ? bufB : bufA);
    run_mlp_layer(X, mlp[l], dst, Xh, mean, invstd, M, stream);
    X = dst;
  }
}

} // namespace

extern "C" void kernel_launch(void* const* d_in, const int* in_sizes, int n_in,
                              void* d_out, int out_size, void* d_ws, size_t ws_size,
                              hipStream_t stream) {
  const int B = 8, N = 4096;
  const float* xyz = (const float*)d_in[0];

  // -------- workspace bump allocator --------
  char* ws = (char*)d_ws;
  size_t off = 0;
  auto allocB = [&](size_t bytes) -> void* {
    void* p = ws + off;
    off += (bytes + 255) & ~(size_t)255;
    return p;
  };
  auto allocF = [&](size_t n) { return (float*)allocB(n * sizeof(float)); };
  auto allocH = [&](size_t n) { return (_Float16*)allocB(n * sizeof(_Float16)); };
  auto allocI = [&](size_t n) { return (int*)allocB(n * sizeof(int)); };

  // -------- parameter walk + weight prep (dict insertion order) --------
  // 23 layers: sa1(3) sa2(3) sa3(3) sa4(3) fp4(2) fp3(2) fp2(2) fp1(3)
  //            head1(1) head2(1, no BN params)
  static const int KO[23][2] = {
      {9, 32},   {32, 32},  {32, 64},                 // sa1
      {67, 64},  {64, 64},  {64, 128},                // sa2
      {131, 128},{128, 128},{128, 256},               // sa3
      {259, 256},{256, 256},{256, 512},               // sa4
      {768, 256},{256, 256},                          // fp4
      {384, 256},{256, 256},                          // fp3
      {320, 256},{256, 128},                          // fp2
      {128, 128},{128, 128},{128, 128},               // fp1
      {128, 128},                                     // head1
      {128, 8}                                        // head2
  };
  Prepped L[23];
  int pi = 1;
  for (int i = 0; i < 23; ++i) {
    const float* W = (const float*)d_in[pi++];
    const float* b = (const float*)d_in[pi++];
    const float* g = nullptr;
    const float* be = nullptr;
    if (i < 22) { g = (const float*)d_in[pi++]; be = (const float*)d_in[pi++]; }
    const int K = KO[i][0], O = KO[i][1];
    const int Kpad = (K + 31) & ~31;
    const int Opad = (O + 15) & ~15;
    _Float16* Wh = allocH((size_t)Opad * Kpad);
    float* bp = allocF(Opad);
    cvt_w_kernel<<<ceil_div(Opad * Kpad, 256), 256, 0, stream>>>(
        W, b, Wh, bp, O, K, Opad, Kpad);
    L[i] = Prepped{Wh, bp, g, be, K, O, Kpad, Opad};
  }

  // -------- activations / scratch --------
  float* xyz_t  = allocF((size_t)B * N * 3);
  float* pts_t  = allocF((size_t)B * N * 6);
  float* l1_xyz = allocF((size_t)B * 1024 * 3);
  float* l2_xyz = allocF((size_t)B * 256 * 3);
  float* l3_xyz = allocF((size_t)B * 64 * 3);
  float* l4_xyz = allocF((size_t)B * 16 * 3);
  float* l1_pts = allocF((size_t)B * 1024 * 64);
  float* l2_pts = allocF((size_t)B * 256 * 128);
  float* l3_pts = allocF((size_t)B * 64 * 256);
  float* l4_pts = allocF((size_t)B * 16 * 512);
  float* fpo4   = allocF((size_t)B * 64 * 256);
  float* fpo3   = allocF((size_t)B * 256 * 256);
  float* fpo2   = allocF((size_t)B * 1024 * 128);
  float* fpo1   = allocF((size_t)B * 4096 * 128);
  float* mean   = allocF(512);
  float* invstd = allocF(512);
  int*   fpsidx = allocI((size_t)B * 1024);
  int*   ballix = allocI((size_t)B * 1024 * 32);
  float* bufA   = allocF((size_t)262144 * 64);   // max M*C (SA1 layer 3)
  float* bufB   = allocF((size_t)262144 * 64);
  _Float16* Xh  = allocH((size_t)262144 * 36);   // max M*Kpad (SA1: 262144x32, +slack)

  // -------- pipeline --------
  split_input_kernel<<<ceil_div(B * N, 256), 256, 0, stream>>>(xyz, xyz_t, pts_t, B, N);

  run_sa(xyz_t,  pts_t,  N,    6,   1024, 0.1f, &L[0], 3, l1_xyz, l1_pts,
         fpsidx, ballix, bufA, bufB, Xh, mean, invstd, B, stream);
  run_sa(l1_xyz, l1_pts, 1024, 64,  256,  0.2f, &L[3], 3, l2_xyz, l2_pts,
         fpsidx, ballix, bufA, bufB, Xh, mean, invstd, B, stream);
  run_sa(l2_xyz, l2_pts, 256,  128, 64,   0.4f, &L[6], 3, l3_xyz, l3_pts,
         fpsidx, ballix, bufA, bufB, Xh, mean, invstd, B, stream);
  run_sa(l3_xyz, l3_pts, 64,   256, 16,   0.8f, &L[9], 3, l4_xyz, l4_pts,
         fpsidx, ballix, bufA, bufB, Xh, mean, invstd, B, stream);

  run_fp(l3_xyz, l4_xyz, l3_pts, 256, l4_pts, 512, 64,   16,   &L[12], 2,
         fpo4, bufA, bufB, Xh, mean, invstd, B, stream);
  run_fp(l2_xyz, l3_xyz, l2_pts, 128, fpo4,   256, 256,  64,   &L[14], 2,
         fpo3, bufA, bufB, Xh, mean, invstd, B, stream);
  run_fp(l1_xyz, l2_xyz, l1_pts, 64,  fpo3,   256, 1024, 256,  &L[16], 2,
         fpo2, bufA, bufB, Xh, mean, invstd, B, stream);
  run_fp(xyz_t,  l1_xyz, nullptr, 0,  fpo2,   128, 4096, 1024, &L[18], 3,
         fpo1, bufA, bufB, Xh, mean, invstd, B, stream);

  // head: conv_bn_relu_1d(128->128), linear(128->8, Opad=16), log_softmax
  const int Mh = B * N;
  run_mlp_layer(fpo1, L[21], bufA, Xh, mean, invstd, Mh, stream, /*bn_relu=*/true);
  run_mlp_layer(bufA, L[22], bufB, Xh, mean, invstd, Mh, stream, /*bn_relu=*/false);
  logsoftmax_kernel<<<ceil_div(Mh, 256), 256, 0, stream>>>(
      bufB, (float*)d_out, B, N, 8, L[22].Opad);
}
